// Phrase_Context_32916629357177
// MI455X (gfx1250) — compile-verified, matrix operand order checked by reading, not run on hardware
//
#include <hip/hip_runtime.h>

// ---------------------------------------------------------------------------
// Types for CDNA5 WMMA (wave32, 16x16x32 bf16 -> f32 accumulate)
// ---------------------------------------------------------------------------
typedef __bf16 bf16_t;
typedef __attribute__((ext_vector_type(16))) __bf16 v16bf;
typedef __attribute__((ext_vector_type(8)))  float  v8f;
typedef __attribute__((ext_vector_type(4)))  int    i32x4;

// Problem constants (from reference)
#define CH   512      // HDIM
#define TT   256      // T
#define BBATCH 16     // B
#define NP   8        // NPHRASE
#define NH   8        // heads
#define DHD  64       // head dim
#define BNR  128      // B*NP
#define RK   32       // RANK
#define NCF  4096     // NP*CH

// GEMM tiling
#define BM   128
#define BNT  64
#define BK   32
#define LSA  40       // LDS row stride (bf16 elems), padded
#define LSB  40

#define FLAG_RELU  1
#define FLAG_ACC   2
#define FLAG_OUTBF 4

// gfx1250 async global->LDS staging (guarded: falls back to sync staging)
#if defined(__gfx1250__) && __has_builtin(__builtin_amdgcn_global_load_async_to_lds_b128)
#define HAVE_ASYNC_LDS 1
#else
#define HAVE_ASYNC_LDS 0
#endif

union FragU { uint4 q[2]; v16bf v; };

__device__ __forceinline__ void wait_async_all() {
#if HAVE_ASYNC_LDS
#if __has_builtin(__builtin_amdgcn_s_wait_asynccnt)
  __builtin_amdgcn_s_wait_asynccnt(0);
#else
  asm volatile("s_wait_asynccnt 0x0" ::: "memory");
#endif
#endif
}

// copy 16B global -> LDS (async on CDNA5), or zero-fill LDS if !valid
__device__ __forceinline__ void stage16(const bf16_t* g, bf16_t* l, bool valid) {
#if HAVE_ASYNC_LDS
  if (valid) {
    __builtin_amdgcn_global_load_async_to_lds_b128(
        (__attribute__((address_space(1))) i32x4*)g,
        (__attribute__((address_space(3))) i32x4*)l, 0, 0);
  } else {
    const uint4 z = {0u, 0u, 0u, 0u};
    *(uint4*)l = z;
  }
#else
  uint4 d = {0u, 0u, 0u, 0u};
  if (valid) d = *(const uint4*)g;
  *(uint4*)l = d;
#endif
}

// ---------------------------------------------------------------------------
// Batched WMMA GEMM:  C[z] = alpha * A[z](MxK) * Bt[z](NxK)^T (+bias) (+C) (relu)
//   z decomposes as z1 = z/zdiv, z2 = z%zdiv with independent strides.
//   A rows may be shifted/windowed (a_row_off / a_rows_valid) for conv taps.
//   Double-buffered LDS; next K-tile staged with GLOBAL_LOAD_ASYNC_TO_LDS_B128
//   while WMMAs consume the current tile.
// ---------------------------------------------------------------------------
__global__ __launch_bounds__(256)
void k_gemm(const bf16_t* __restrict__ A, const bf16_t* __restrict__ Bt,
            const float* __restrict__ bias, void* __restrict__ Cout,
            int M, int N, int K, int lda, int ldb, int ldc,
            int zdiv, long azs1, long azs2, long bzs1, long bzs2,
            long czs1, long czs2, float alpha, int flags,
            int a_row_off, int a_rows_valid)
{
  __shared__ __align__(16) bf16_t As[2][BM * LSA];
  __shared__ __align__(16) bf16_t Bs[2][BNT * LSB];

  const int z  = blockIdx.z;
  const int z1 = z / zdiv;
  const int z2 = z - z1 * zdiv;
  A  += (long)z1 * azs1 + (long)z2 * azs2;
  Bt += (long)z1 * bzs1 + (long)z2 * bzs2;
  const long coff = (long)z1 * czs1 + (long)z2 * czs2;

  const int bm   = blockIdx.y * BM;
  const int bn   = blockIdx.x * BNT;
  const int tid  = threadIdx.x;
  const int lane = tid & 31;
  const int wave = tid >> 5;          // 0..7
  const int wm   = wave >> 1;         // 0..3 : 32 rows each
  const int wn   = wave & 1;          // 0..1 : 32 cols each
  const int lm   = lane & 15;
  const int kh   = lane >> 4;         // K-half / M-half selector

  const v8f vzero = {0.f,0.f,0.f,0.f,0.f,0.f,0.f,0.f};
  v8f acc[2][2];
  acc[0][0] = vzero; acc[0][1] = vzero; acc[1][0] = vzero; acc[1][1] = vzero;

  // staging lambdas: A tile = BM x BK (2 x 16B per thread), B tile = BNT x BK
  auto stageA = [&](int buf, int k0) {
    #pragma unroll
    for (int i = 0; i < 2; i++) {
      int idx = tid + i * 256;
      int row = idx >> 2, seg = idx & 3;
      int gm  = bm + row;
      int sr  = gm + a_row_off;
      bool v  = (gm < M) && (sr >= 0) && (sr < a_rows_valid);
      stage16(A + (long)sr * lda + k0 + seg * 8,
              &As[buf][row * LSA + seg * 8], v);
    }
  };
  auto stageB = [&](int buf, int k0) {
    int row = tid >> 2, seg = tid & 3;
    int gn  = bn + row;
    stage16(Bt + (long)gn * ldb + k0 + seg * 8,
            &Bs[buf][row * LSB + seg * 8], gn < N);
  };

  stageA(0, 0);
  stageB(0, 0);
  wait_async_all();
  __syncthreads();

  int buf = 0;
  for (int k0 = 0; k0 < K; k0 += BK) {
    const int kn = k0 + BK;
    if (kn < K) {                       // issue next tile into other buffer
      stageA(buf ^ 1, kn);
      stageB(buf ^ 1, kn);
    }
    if (k0 + 2 * BK < K) {              // keep L2 warm two tiles ahead
      int pr = bm + (tid >> 1);
      int sr = pr + a_row_off;
      if (pr < M && sr >= 0 && sr < a_rows_valid)
        __builtin_prefetch(A + (long)sr * lda + k0 + 2 * BK);
    }

    // fragments: lane lm = row/col, kh selects K-half per ISA VGPR layout
    FragU af[2], bfr[2];
    #pragma unroll
    for (int i = 0; i < 2; i++) {
      const bf16_t* p = &As[buf][(wm * 32 + i * 16 + lm) * LSA + kh * 8];
      af[i].q[0] = *(const uint4*)(p);
      af[i].q[1] = *(const uint4*)(p + 16);
    }
    #pragma unroll
    for (int j = 0; j < 2; j++) {
      const bf16_t* p = &Bs[buf][(wn * 32 + j * 16 + lm) * LSB + kh * 8];
      bfr[j].q[0] = *(const uint4*)(p);
      bfr[j].q[1] = *(const uint4*)(p + 16);
    }
    #pragma unroll
    for (int i = 0; i < 2; i++)
      #pragma unroll
      for (int j = 0; j < 2; j++)
        acc[i][j] = __builtin_amdgcn_wmma_f32_16x16x32_bf16(
            false, af[i].v, false, bfr[j].v, (short)0, acc[i][j], false, false);

    wait_async_all();                   // next-tile async copies landed
    __syncthreads();                    // everyone done reading buf
    buf ^= 1;
  }

  // ---- epilogue: bias / alpha / accumulate / relu / f32-or-bf16 store -----
  float* Cf = (float*)Cout + coff;
  bf16_t* Cb = (bf16_t*)Cout + coff;
  #pragma unroll
  for (int i = 0; i < 2; i++) {
    #pragma unroll
    for (int j = 0; j < 2; j++) {
      int gn = bn + wn * 32 + j * 16 + lm;
      if (gn >= N) continue;
      float bv = bias ? bias[gn] : 0.f;
      #pragma unroll
      for (int r = 0; r < 8; r++) {
        int gm = bm + wm * 32 + i * 16 + kh * 8 + r;
        if (gm >= M) continue;
        long off = (long)gm * ldc + gn;
        float v = alpha * acc[i][j][r] + bv;
        if (flags & FLAG_ACC)  v += Cf[off];
        if (flags & FLAG_RELU) v = fmaxf(v, 0.f);
        if (flags & FLAG_OUTBF) Cb[off] = (bf16_t)v;
        else                    Cf[off] = v;
      }
    }
  }
}

// ---------------------------------------------------------------------------
// Elementwise / reduction kernels
// ---------------------------------------------------------------------------
__device__ __forceinline__ float blk_sum(float v, float* sbuf) {
  int t = threadIdx.x;
  sbuf[t] = v; __syncthreads();
  #pragma unroll
  for (int k = 128; k > 0; k >>= 1) {
    if (t < k) sbuf[t] += sbuf[t + k];
    __syncthreads();
  }
  float r = sbuf[0]; __syncthreads();
  return r;
}

__global__ void k_cvt_bf16(const float* __restrict__ s, bf16_t* __restrict__ d, long n) {
  long i = (long)blockIdx.x * 256 + threadIdx.x;
  if (i < n) d[i] = (bf16_t)s[i];
}

// LayerNorm over C=512 (optional residual, relu, dual f32/bf16 output)
__global__ __launch_bounds__(256)
void k_ln(const float* __restrict__ x, const float* __restrict__ res,
          const float* __restrict__ g, const float* __restrict__ beta,
          float* __restrict__ yf, bf16_t* __restrict__ yb, int relu)
{
  __shared__ float sbuf[256];
  long row = blockIdx.x;
  const float* xr = x + row * CH;
  const float* rr = res ? res + row * CH : nullptr;
  float vals[2], s = 0.f, s2 = 0.f;
  #pragma unroll
  for (int k = 0; k < 2; k++) {
    int c = threadIdx.x + k * 256;
    float t = xr[c] + (rr ? rr[c] : 0.f);
    vals[k] = t; s += t; s2 += t * t;
  }
  s = blk_sum(s, sbuf); s2 = blk_sum(s2, sbuf);
  float mean = s * (1.f / CH);
  float inv  = rsqrtf(s2 * (1.f / CH) - mean * mean + 1e-5f);
  #pragma unroll
  for (int k = 0; k < 2; k++) {
    int c = threadIdx.x + k * 256;
    float v = (vals[k] - mean) * inv * g[c] + beta[c];
    if (relu) v = fmaxf(v, 0.f);
    if (yf) yf[row * CH + c] = v;
    if (yb) yb[row * CH + c] = (bf16_t)v;
  }
}

// h = LN(relu'd x1[b,n,:] * relu'd x2[b,t,:]) -> bf16   (row = (b*NP+n)*TT + t)
__global__ __launch_bounds__(256)
void k_had_ln(const float* __restrict__ x1, const float* __restrict__ x2,
              const float* __restrict__ g, const float* __restrict__ beta,
              bf16_t* __restrict__ out)
{
  __shared__ float sbuf[256];
  long row = blockIdx.x;
  int  t   = (int)(row & (TT - 1));
  long rn  = row >> 8;          // b*NP+n
  long b   = rn >> 3;
  const float* p1 = x1 + rn * CH;
  const float* p2 = x2 + (b * TT + t) * CH;
  float vals[2], s = 0.f, s2 = 0.f;
  #pragma unroll
  for (int k = 0; k < 2; k++) {
    int c = threadIdx.x + k * 256;
    float v = p1[c] * p2[c];
    vals[k] = v; s += v; s2 += v * v;
  }
  s = blk_sum(s, sbuf); s2 = blk_sum(s2, sbuf);
  float mean = s * (1.f / CH);
  float inv  = rsqrtf(s2 * (1.f / CH) - mean * mean + 1e-5f);
  #pragma unroll
  for (int k = 0; k < 2; k++) {
    int c = threadIdx.x + k * 256;
    out[row * CH + c] = (bf16_t)((vals[k] - mean) * inv * g[c] + beta[c]);
  }
}

// softmax over rows of length TT=256, bf16 output (mask is all-valid)
__global__ __launch_bounds__(256)
void k_softmax(const float* __restrict__ s, bf16_t* __restrict__ o)
{
  __shared__ float sbuf[256];
  long row = blockIdx.x;
  int  t   = threadIdx.x;
  float v = s[row * TT + t];
  sbuf[t] = v; __syncthreads();
  #pragma unroll
  for (int k = 128; k > 0; k >>= 1) {
    if (t < k) sbuf[t] = fmaxf(sbuf[t], sbuf[t + k]);
    __syncthreads();
  }
  float m = sbuf[0]; __syncthreads();
  float e = __expf(v - m);
  sbuf[t] = e; __syncthreads();
  #pragma unroll
  for (int k = 128; k > 0; k >>= 1) {
    if (t < k) sbuf[t] += sbuf[t + k];
    __syncthreads();
  }
  o[row * TT + t] = (bf16_t)(e / sbuf[0]);
}

// DETR sine positional embedding add: xf = ctx + pos, xb = bf16(xf)
__global__ void k_pos(const float* __restrict__ ctx, float* __restrict__ xf,
                      bf16_t* __restrict__ xb)
{
  long i = (long)blockIdx.x * 256 + threadIdx.x;
  if (i >= (long)BNR * TT * CH) return;
  int c = (int)(i & (CH - 1));
  int t = (int)((i >> 9) & (TT - 1));
  float e  = (float)(t + 1) * (6.28318530718f / (256.0f + 1e-6f));
  float ex = (float)((c >> 1) << 1) / (float)CH;
  float arg = e / __powf(10000.0f, ex);
  float pv  = (c & 1) ? __cosf(arg) : __sinf(arg);
  float v = ctx[i] + pv;
  xf[i] = v; xb[i] = (bf16_t)v;
}

// vp [BN,T,C] -> vt [BN,H,DH,T] bf16 (head-transposed for K-minor GEMM)
__global__ void k_vt(const float* __restrict__ vp, bf16_t* __restrict__ vt)
{
  long i = (long)blockIdx.x * 256 + threadIdx.x;
  if (i >= (long)BNR * TT * CH) return;
  int c  = (int)(i & (CH - 1));
  int t  = (int)((i >> 9) & (TT - 1));
  long bn = i >> 17;
  int h = c >> 6, d = c & 63;
  vt[(((bn * NH + h) * DHD) + d) * TT + t] = (bf16_t)vp[i];
}

// refined x [B,N,T,C] (f32) -> ctx_flat [B,T,N*C] bf16
__global__ void k_btnc(const float* __restrict__ x, bf16_t* __restrict__ out)
{
  long i = (long)blockIdx.x * 256 + threadIdx.x;
  if (i >= (long)BBATCH * TT * NCF) return;
  int c = (int)(i & (CH - 1));
  int n = (int)((i >> 9) & (NP - 1));
  int t = (int)((i >> 12) & (TT - 1));
  long b = i >> 20;
  out[i] = (bf16_t)x[(((b * NP + n) * TT) + t) * CH + c];
}

// comb[b,n] = [phrase_slot[b,n,:], eos_slot[b,:]] -> bf16 [BNR, 2C]
__global__ void k_comb(const float* __restrict__ ph, const float* __restrict__ eos,
                       bf16_t* __restrict__ comb)
{
  long i = (long)blockIdx.x * 256 + threadIdx.x;
  if (i >= (long)BNR * 2 * CH) return;
  int  c2 = (int)(i & (2 * CH - 1));
  long rn = i >> 10;
  long b  = rn >> 3;
  float v = (c2 < CH) ? ph[rn * CH + c2] : eos[b * CH + (c2 - CH)];
  comb[i] = (bf16_t)v;
}

// kern[r,d,kk] -> kernT[seg][d][r] bf16; seg order: k1:0 | k3:0..2 | k5:0..4
__global__ void k_kern_t(const float* __restrict__ k1, const float* __restrict__ k3,
                         const float* __restrict__ k5, bf16_t* __restrict__ kt)
{
  long i = (long)blockIdx.x * 256 + threadIdx.x;
  if (i >= (long)9 * CH * RK) return;
  int r   = (int)(i & 31);
  int d   = (int)((i >> 5) & (CH - 1));
  int seg = (int)(i >> 14);
  const float* src; int kw, kk;
  if (seg == 0)      { src = k1; kw = 1; kk = 0; }
  else if (seg < 4)  { src = k3; kw = 3; kk = seg - 1; }
  else               { src = k5; kw = 5; kk = seg - 4; }
  kt[i] = (bf16_t)src[(r * CH + d) * kw + kk];
}

// ---------------------------------------------------------------------------
// Host-side helpers
// ---------------------------------------------------------------------------
struct GemmCfg {
  int M, N, K, lda, ldb, ldc;
  int batches, zdiv;
  long azs1, azs2, bzs1, bzs2, czs1, czs2;
  float alpha; int flags; int arow_off; int arows;
  GemmCfg(int M_, int N_, int K_, int lda_, int ldb_, int ldc_)
      : M(M_), N(N_), K(K_), lda(lda_), ldb(ldb_), ldc(ldc_),
        batches(1), zdiv(1), azs1(0), azs2(0), bzs1(0), bzs2(0),
        czs1(0), czs2(0), alpha(1.f), flags(0), arow_off(0), arows(-1) {}
};

static void gemm(hipStream_t st, const bf16_t* A, const bf16_t* Bt,
                 const float* bias, void* C, GemmCfg c)
{
  if (c.arows < 0) c.arows = c.M;
  dim3 g((c.N + BNT - 1) / BNT, (c.M + BM - 1) / BM, c.batches);
  k_gemm<<<g, 256, 0, st>>>(A, Bt, bias, C, c.M, c.N, c.K, c.lda, c.ldb, c.ldc,
                            c.zdiv, c.azs1, c.azs2, c.bzs1, c.bzs2, c.czs1, c.czs2,
                            c.alpha, c.flags, c.arow_off, c.arows);
}

extern "C" void kernel_launch(void* const* d_in, const int* in_sizes, int n_in,
                              void* d_out, int out_size, void* d_ws, size_t ws_size,
                              hipStream_t stream)
{
  auto in = [&](int i) { return (const float*)d_in[i]; };
  const float* phrase = in(0);
  const float* vid    = in(1);
  // d_in[2] = vid_mask (all-true, unused)
  const float* eos    = in(3);
  int p = 4;
  const float* hp_w1 = in(p++); const float* hp_b1 = in(p++);
  const float* hp_w2 = in(p++); const float* hp_b2 = in(p++);
  const float* hp_w3 = in(p++); const float* hp_b3 = in(p++);
  const float* hp_g  = in(p++); const float* hp_be = in(p++);
  const float* hp_g1 = in(p++); const float* hp_be1= in(p++);
  struct LayerW { const float *wq,*bq,*wk,*bk,*wv,*bv,*inw,*inb,*outw,*outb,
                              *attg,*attb,*fcw,*fcb,*ntg,*ntb; } L[2];
  for (int l = 0; l < 2; l++) {
    L[l].wq  = in(p++); L[l].bq  = in(p++);
    L[l].wk  = in(p++); L[l].bk  = in(p++);
    L[l].wv  = in(p++); L[l].bv  = in(p++);
    L[l].inw = in(p++); L[l].inb = in(p++);
    L[l].outw= in(p++); L[l].outb= in(p++);
    L[l].attg= in(p++); L[l].attb= in(p++);
    L[l].fcw = in(p++); L[l].fcb = in(p++);
    L[l].ntg = in(p++); L[l].ntb = in(p++);
  }
  const float* cp_w1 = in(p++); const float* cp_b1 = in(p++);
  const float* cp_w2 = in(p++); const float* cp_b2 = in(p++);
  const float* kern1 = in(p++); const float* kern3 = in(p++); const float* kern5 = in(p++);
  const float* lo_w  = in(p++); const float* lo_b  = in(p++);
  const float* lc_g  = in(p++); const float* lc_b  = in(p++);

  // outputs: (agg [B,T,C], emb [B,N,T,C], refine [B,N,T,C]) concatenated
  float* d_agg    = (float*)d_out;
  float* d_emb    = d_agg + (long)BBATCH * TT * CH;
  float* d_refine = d_emb + (long)BBATCH * NP * TT * CH;

  // workspace bump allocator (deterministic)
  char* wsb = (char*)d_ws;
  size_t off = 0;
  auto alloc = [&](size_t bytes) -> void* {
    void* q = wsb + off;
    off = (off + bytes + 255) & ~(size_t)255;
    return q;
  };
  const long ROWS = (long)BNR * TT;           // 32768
  const long BIG  = ROWS * CH;                // 16,777,216

  // bf16 weight mirrors
  bf16_t* w1b  = (bf16_t*)alloc(CH * CH * 2);
  bf16_t* w2b  = (bf16_t*)alloc(CH * CH * 2);
  bf16_t* w3b  = (bf16_t*)alloc(CH * CH * 2);
  bf16_t* lwb[2][6];
  for (int l = 0; l < 2; l++) {
    lwb[l][0] = (bf16_t*)alloc(CH * CH * 2);       // wq
    lwb[l][1] = (bf16_t*)alloc(CH * CH * 2);       // wk
    lwb[l][2] = (bf16_t*)alloc(CH * CH * 2);       // wv
    lwb[l][3] = (bf16_t*)alloc(3 * CH * CH * 2);   // in_w
    lwb[l][4] = (bf16_t*)alloc(CH * CH * 2);       // out_w
    lwb[l][5] = (bf16_t*)alloc(CH * CH * 2);       // fc_w
  }
  bf16_t* cpw1b = (bf16_t*)alloc((long)2048 * 1024 * 2);
  bf16_t* cpw2b = (bf16_t*)alloc((long)16384 * 2048 * 2);
  bf16_t* lowb  = (bf16_t*)alloc((long)CH * 1536 * 2);
  bf16_t* kernT = (bf16_t*)alloc((long)9 * CH * RK * 2);

  // activations
  bf16_t* psb  = (bf16_t*)alloc((long)BNR * CH * 2);
  bf16_t* vfb  = (bf16_t*)alloc((long)BBATCH * TT * CH * 2);
  float*  x1f  = (float*)alloc((long)BNR * CH * 4);
  float*  x2f  = (float*)alloc((long)BBATCH * TT * CH * 4);
  bf16_t* hb   = (bf16_t*)alloc(BIG * 2);
  float*  tmpf = (float*)alloc(BIG * 4);
  float*  xf   = (float*)alloc(BIG * 4);
  bf16_t* xb   = (bf16_t*)alloc(BIG * 2);
  float *qf = (float*)alloc(BIG * 4), *kf = (float*)alloc(BIG * 4), *vf = (float*)alloc(BIG * 4);
  bf16_t *qb = (bf16_t*)alloc(BIG * 2), *kb = (bf16_t*)alloc(BIG * 2), *vb = (bf16_t*)alloc(BIG * 2);
  float *qpf = (float*)alloc(BIG * 4), *kpf = (float*)alloc(BIG * 4), *vpf = (float*)alloc(BIG * 4);
  bf16_t *qpb = (bf16_t*)alloc(BIG * 2), *kpb = (bf16_t*)alloc(BIG * 2);
  const long SCN = (long)BNR * NH * TT * TT;  // 67,108,864
  float*  scores = (float*)alloc(SCN * 4);
  bf16_t* attnb  = (bf16_t*)alloc(SCN * 2);
  bf16_t* vtb    = (bf16_t*)alloc(BIG * 2);
  float*  aout   = (float*)alloc(BIG * 4);
  bf16_t* aob    = (bf16_t*)alloc(BIG * 2);
  float*  updf   = (float*)alloc(BIG * 4);
  float*  tf     = (float*)alloc(BIG * 4);
  bf16_t* ctxfl  = (bf16_t*)alloc(BIG * 2);
  bf16_t* combb  = (bf16_t*)alloc((long)BNR * 2 * CH * 2);
  float*  hf     = (float*)alloc((long)BNR * 2048 * 4);
  bf16_t* hfb    = (bf16_t*)alloc((long)BNR * 2048 * 2);
  float*  projf  = (float*)alloc((long)BNR * 16384 * 4);
  bf16_t* projb  = (bf16_t*)alloc((long)BNR * 16384 * 2);
  const long DSEG = (long)CH * NCF;           // 2,097,152 per (b,seg) tile
  bf16_t* dynT   = (bf16_t*)alloc((long)BBATCH * 9 * DSEG * 2);
  float*  feat   = (float*)alloc((long)BBATCH * TT * 1536 * 4);
  bf16_t* featb  = (bf16_t*)alloc((long)BBATCH * TT * 1536 * 2);
  float*  outf   = (float*)alloc((long)BBATCH * TT * CH * 4);
  (void)ws_size; (void)in_sizes; (void)n_in; (void)out_size;

  auto cvt = [&](const float* s, bf16_t* d, long n) {
    k_cvt_bf16<<<(unsigned)((n + 255) / 256), 256, 0, stream>>>(s, d, n);
  };

  // ---- weight conversions -------------------------------------------------
  cvt(hp_w1, w1b, CH * CH); cvt(hp_w2, w2b, CH * CH); cvt(hp_w3, w3b, CH * CH);
  for (int l = 0; l < 2; l++) {
    cvt(L[l].wq,  lwb[l][0], CH * CH);
    cvt(L[l].wk,  lwb[l][1], CH * CH);
    cvt(L[l].wv,  lwb[l][2], CH * CH);
    cvt(L[l].inw, lwb[l][3], 3L * CH * CH);
    cvt(L[l].outw,lwb[l][4], CH * CH);
    cvt(L[l].fcw, lwb[l][5], CH * CH);
  }
  cvt(cp_w1, cpw1b, 2048L * 1024);
  cvt(cp_w2, cpw2b, 16384L * 2048);
  cvt(lo_w,  lowb,  (long)CH * 1536);
  k_kern_t<<<(9 * CH * RK + 255) / 256, 256, 0, stream>>>(kern1, kern3, kern5, kernT);

  // ---- HadamardProduct fusion --------------------------------------------
  cvt(phrase, psb, (long)BNR * CH);
  cvt(vid,    vfb, (long)BBATCH * TT * CH);
  { GemmCfg c(BNR, CH, CH, CH, CH, CH); c.flags = FLAG_RELU;
    gemm(stream, psb, w1b, hp_b1, x1f, c); }
  { GemmCfg c(BBATCH * TT, CH, CH, CH, CH, CH); c.flags = FLAG_RELU;
    gemm(stream, vfb, w2b, hp_b2, x2f, c); }
  k_had_ln<<<(unsigned)ROWS, 256, 0, stream>>>(x1f, x2f, hp_g, hp_be, hb);
  { GemmCfg c((int)ROWS, CH, CH, CH, CH, CH);
    gemm(stream, hb, w3b, hp_b3, tmpf, c); }
  k_ln<<<(unsigned)ROWS, 256, 0, stream>>>(tmpf, nullptr, hp_g1, hp_be1, d_emb, nullptr, 1);
  k_pos<<<(unsigned)((BIG + 255) / 256), 256, 0, stream>>>(d_emb, xf, xb);

  // ---- transformer layers -------------------------------------------------
  for (int l = 0; l < 2; l++) {
    { GemmCfg c((int)ROWS, CH, CH, CH, CH, CH); gemm(stream, xb, lwb[l][0], L[l].bq, qf, c); }
    { GemmCfg c((int)ROWS, CH, CH, CH, CH, CH); gemm(stream, xb, lwb[l][1], L[l].bk, kf, c); }
    { GemmCfg c((int)ROWS, CH, CH, CH, CH, CH); gemm(stream, xb, lwb[l][2], L[l].bv, vf, c); }
    cvt(qf, qb, BIG); cvt(kf, kb, BIG); cvt(vf, vb, BIG);
    { GemmCfg c((int)ROWS, CH, CH, CH, CH, CH);
      gemm(stream, qb, lwb[l][3],               L[l].inb,          qpf, c); }
    { GemmCfg c((int)ROWS, CH, CH, CH, CH, CH);
      gemm(stream, kb, lwb[l][3] + CH * CH,     L[l].inb + CH,     kpf, c); }
    { GemmCfg c((int)ROWS, CH, CH, CH, CH, CH);
      gemm(stream, vb, lwb[l][3] + 2 * CH * CH, L[l].inb + 2 * CH, vpf, c); }
    cvt(qpf, qpb, BIG); cvt(kpf, kpb, BIG);
    // scores[bn,h] = 0.125 * qp_h (TxDH) . kp_h (TxDH)^T
    { GemmCfg c(TT, TT, DHD, CH, CH, TT);
      c.batches = BNR * NH; c.zdiv = NH;
      c.azs1 = (long)TT * CH; c.azs2 = DHD;
      c.bzs1 = (long)TT * CH; c.bzs2 = DHD;
      c.czs1 = (long)NH * TT * TT; c.czs2 = (long)TT * TT;
      c.alpha = 0.125f;
      gemm(stream, qpb, kpb, nullptr, scores, c); }
    k_softmax<<<(unsigned)(BNR * NH * TT), 256, 0, stream>>>(scores, attnb);
    k_vt<<<(unsigned)((BIG + 255) / 256), 256, 0, stream>>>(vpf, vtb);
    // out[bn,:,h*DH..] = attn (TxT) . vt_h (DHxT)^T
    { GemmCfg c(TT, DHD, TT, TT, TT, CH);
      c.batches = BNR * NH; c.zdiv = NH;
      c.azs1 = (long)NH * TT * TT; c.azs2 = (long)TT * TT;
      c.bzs1 = (long)NH * DHD * TT; c.bzs2 = (long)DHD * TT;
      c.czs1 = (long)TT * CH; c.czs2 = DHD;
      gemm(stream, attnb, vtb, nullptr, aout, c); }
    cvt(aout, aob, BIG);
    { GemmCfg c((int)ROWS, CH, CH, CH, CH, CH);
      gemm(stream, aob, lwb[l][4], L[l].outb, updf, c); }
    k_ln<<<(unsigned)ROWS, 256, 0, stream>>>(xf, updf, L[l].attg, L[l].attb, xf, xb, 0);
    { GemmCfg c((int)ROWS, CH, CH, CH, CH, CH); c.flags = FLAG_RELU;
      gemm(stream, xb, lwb[l][5], L[l].fcb, tf, c); }
    float* ydst = (l == 1) ? d_refine : xf;
    k_ln<<<(unsigned)ROWS, 256, 0, stream>>>(xf, tf, L[l].ntg, L[l].ntb, ydst, xb, 0);
  }

  // ---- low-rank dynamic conv ---------------------------------------------
  k_btnc<<<(unsigned)((BIG + 255) / 256), 256, 0, stream>>>(d_refine, ctxfl);
  k_comb<<<(unsigned)((BNR * 2 * CH + 255) / 256), 256, 0, stream>>>(phrase, eos, combb);
  { GemmCfg c(BNR, 2048, 1024, 1024, 1024, 2048); c.flags = FLAG_RELU;
    gemm(stream, combb, cpw1b, cp_b1, hf, c); }
  cvt(hf, hfb, (long)BNR * 2048);
  { GemmCfg c(BNR, 16384, 2048, 2048, 2048, 16384);
    gemm(stream, hfb, cpw2b, cp_b2, projf, c); }
  cvt(projf, projb, (long)BNR * 16384);
  // dynT[b][seg] (C x NC, bf16) = kernT[seg] (C x R) . proj_b ((n,c) x R)^T
  { GemmCfg c(CH, NCF, RK, RK, RK, NCF);
    c.batches = BBATCH * 9; c.zdiv = 9;
    c.azs2 = (long)CH * RK;                 // seg stride in kernT
    c.bzs1 = (long)NP * 16384;              // per-b proj stride
    c.czs1 = 9L * DSEG; c.czs2 = DSEG;
    c.flags = FLAG_OUTBF;
    gemm(stream, kernT, projb, nullptr, dynT, c); }
  // conv taps: feat[:, :, cb*512..] += ctx_flat(shifted) . dynT[seg]^T
  {
    int seg = 0;
    const int kws[3] = {1, 3, 5};
    for (int ki = 0; ki < 3; ki++) {
      int kw = kws[ki];
      for (int kk = 0; kk < kw; kk++) {
        GemmCfg c(TT, CH, NCF, NCF, NCF, 1536);
        c.batches = BBATCH; c.zdiv = 1;
        c.azs1 = (long)TT * NCF;
        c.bzs1 = 9L * DSEG;
        c.czs1 = (long)TT * 1536;
        c.flags = (kk > 0) ? FLAG_ACC : 0;
        c.arow_off = kk - kw / 2; c.arows = TT;
        gemm(stream, ctxfl, dynT + (long)seg * DSEG, nullptr,
             feat + ki * CH, c);
        seg++;
      }
    }
  }
  cvt(feat, featb, (long)BBATCH * TT * 1536);
  { GemmCfg c(BBATCH * TT, CH, 1536, 1536, 1536, CH);
    gemm(stream, featb, lowb, lo_b, outf, c); }
  k_ln<<<(unsigned)(BBATCH * TT), 256, 0, stream>>>(outf, nullptr, lc_g, lc_b, d_agg, nullptr, 1);
}
